// SelfAttention_90305982365718
// MI455X (gfx1250) — matrix operand order repaired and checked by reading, not compiled, not run on hardware
//
#include <hip/hip_runtime.h>

// ---------------------------------------------------------------------------
// Self-attention for MI455X (gfx1250, wave32, WMMA f32_16x16x32_f16)
// N=4, S=2048, E=1024, H=16, D=64
// ---------------------------------------------------------------------------

typedef __attribute__((ext_vector_type(16))) _Float16 v16h;
typedef __attribute__((ext_vector_type(8)))  _Float16 v8h;
typedef __attribute__((ext_vector_type(8)))  float    v8f;
typedef __attribute__((ext_vector_type(4)))  float    v4f;

constexpr int N_B   = 4;
constexpr int S_LEN = 2048;
constexpr int E_DIM = 1024;
constexpr int H_N   = 16;
constexpr int D_H   = 64;

static __device__ __forceinline__ v8f wmma_f16(v16h a, v16h b, v8f c) {
    return __builtin_amdgcn_wmma_f32_16x16x32_f16(
        /*neg_a=*/false, a, /*neg_b=*/false, b,
        /*c_mod=*/(short)0, c, /*reuse_a=*/false, /*reuse_b=*/false);
}

// ---------------------------------------------------------------------------
// Stage 0: weight conversion f32 -> f16
// ---------------------------------------------------------------------------
__global__ void cvt_f32_f16_kernel(const float* __restrict__ src,
                                   _Float16* __restrict__ dst, int n) {
    int i = blockIdx.x * 256 + threadIdx.x;
    if (i < n) dst[i] = (_Float16)src[i];
}

// ---------------------------------------------------------------------------
// Stage 1: per-head projection  out = x @ W^T per (n,s,h) row.
// GEMM: M = N*S*H rows (64 f32 contiguous each), N = 64, K = 64.
// TRANS=false: out[n,h,s,d]   (Q,K — rows contiguous in d)
// TRANS=true : out[n,h,d,s]   (V — columns contiguous in s, feeds P@V B-frags)
// Non-transposed epilogue staged through wave-private LDS -> b128 stores.
// ---------------------------------------------------------------------------
template <bool TRANS>
__global__ __launch_bounds__(256) void proj_kernel(
        const float* __restrict__ in, const _Float16* __restrict__ W16,
        _Float16* __restrict__ out) {
    __shared__ _Float16 stage[8 * 16 * 64];   // 16 KB, one slab per wave
    const int wave = threadIdx.x >> 5;
    const int lane = threadIdx.x & 31;
    const int hl   = lane >> 4;      // lane-half (0 or 1)
    const int ln   = lane & 15;
    const int rowBase = blockIdx.x * 128 + wave * 16;
    _Float16* slab = stage + wave * 16 * 64;

    // A fragments: row = rowBase + ln, contiguous K-runs (k = 16*hl + i), same order as B
    const int r = rowBase + ln;
    const int n = r / (S_LEN * H_N);
    const int s = (r / H_N) % S_LEN;
    const int h = r % H_N;
    const float* src = in + ((size_t)(n * S_LEN + s)) * E_DIM + h * D_H;

    v16h a[2];
    for (int kc = 0; kc < 2; ++kc) {
        const int base = kc * 32 + hl * 16;
        for (int i = 0; i < 8; ++i) {
            a[kc][i]     = (_Float16)src[base + i];
            a[kc][i + 8] = (_Float16)src[base + 8 + i];
        }
    }

    for (int j = 0; j < 4; ++j) {
        v8f c; for (int i = 0; i < 8; ++i) c[i] = 0.0f;
        for (int kc = 0; kc < 2; ++kc) {
            // B frag: column e = j*16+ln holds W row e; 16 contiguous K values
            const _Float16* wp = W16 + (size_t)(j * 16 + ln) * D_H + kc * 32 + hl * 16;
            v16h b = *(const v16h*)wp;
            c = wmma_f16(a[kc], b, c);
        }
        if (TRANS) {
            // transposed global layout: inherently scattered, store directly
            for (int rr = 0; rr < 8; ++rr) {
                const int row = rowBase + rr + 8 * hl;
                const int n2 = row / (S_LEN * H_N);
                const int s2 = (row / H_N) % S_LEN;
                const int h2 = row % H_N;
                out[((size_t)(n2 * H_N + h2) * D_H + (j * 16 + ln)) * S_LEN + s2] =
                    (_Float16)c[rr];
            }
        } else {
            for (int rr = 0; rr < 8; ++rr)
                slab[(rr + 8 * hl) * 64 + j * 16 + ln] = (_Float16)c[rr];
        }
    }

    if (!TRANS) {
        // coalesced writeback: 16 rows x 128B, 16B chunks per lane
        for (int t = lane; t < 128; t += 32) {
            const int rowIdx = t >> 3;
            const int piece  = t & 7;
            const int row = rowBase + rowIdx;
            const int n2 = row / (S_LEN * H_N);
            const int s2 = (row / H_N) % S_LEN;
            const int h2 = row % H_N;
            *(v8h*)(out + ((size_t)(n2 * H_N + h2) * S_LEN + s2) * D_H + piece * 8) =
                *(const v8h*)(slab + rowIdx * 64 + piece * 8);
        }
    }
}

// ---------------------------------------------------------------------------
// Stage 2: fused flash attention per (n,h). Workgroup = 64 query rows,
// 4 waves x 16 rows, barrier-free: K read row-major, V read from transposed
// global layout (contiguous B-frags), P re-fragmented through wave-private
// LDS (b128 reads). Online softmax in f32 registers.
// ---------------------------------------------------------------------------
__global__ __launch_bounds__(128) void attn_kernel(
        const _Float16* __restrict__ qp, const _Float16* __restrict__ kp,
        const _Float16* __restrict__ vpT, const int* __restrict__ mask,
        _Float16* __restrict__ attn_out) {
    __shared__ _Float16 Pt[4 * 16 * 64];    // 8 KB probs (wave-private slabs)

    const int wave = threadIdx.x >> 5;
    const int lane = threadIdx.x & 31;
    const int hl   = lane >> 4;
    const int ln   = lane & 15;
    const int nh   = blockIdx.y;            // n*H + h
    const int n    = nh >> 4;
    const int h    = nh & 15;
    const int qbase = blockIdx.x * 64 + wave * 16;

    const _Float16* Kbase  = kp  + (size_t)nh * S_LEN * D_H;
    const _Float16* VTbase = vpT + (size_t)nh * D_H * S_LEN;
    const int* mrow = mask + n * S_LEN;
    _Float16* Pw = Pt + wave * 16 * 64;

    // Q fragments for this wave's 16 rows (held in registers for whole loop)
    v16h aQ[2];
    {
        const _Float16* qrow = qp + ((size_t)nh * S_LEN + (qbase + ln)) * D_H;
        for (int kc = 0; kc < 2; ++kc) {
            const int base = kc * 32 + hl * 16;
            v8h lo = *(const v8h*)(qrow + base);
            v8h hi = *(const v8h*)(qrow + base + 8);
            for (int i = 0; i < 8; ++i) { aQ[kc][i] = lo[i]; aQ[kc][i + 8] = hi[i]; }
        }
    }

    float m_r[8], l_r[8];
    v8f accO[4];
    for (int rr = 0; rr < 8; ++rr) { m_r[rr] = -3.0e38f; l_r[rr] = 0.0f; }
    for (int j = 0; j < 4; ++j) for (int rr = 0; rr < 8; ++rr) accO[j][rr] = 0.0f;

    const float scale = 0.03125f;   // 1/sqrt(E) = 1/32
    constexpr int NBLK = S_LEN / 64;

    for (int kb = 0; kb < NBLK; ++kb) {
        // prefetch next key block of K and V-transposed (global_prefetch_b8)
        if (kb + 1 < NBLK) {
            const int rowp = threadIdx.x >> 1;          // 0..63
            const int hb   = (threadIdx.x & 1) * 32;    // half-row
            __builtin_prefetch(Kbase + (size_t)((kb + 1) * 64 + rowp) * D_H + hb, 0, 1);
            __builtin_prefetch(VTbase + (size_t)rowp * S_LEN + (kb + 1) * 64 + hb, 0, 1);
        }

        // ---- scores S = Q @ K^T for this wave's 16x64 strip --------------
        v8f sc[4];
        for (int j = 0; j < 4; ++j) {
            v8f c; for (int i = 0; i < 8; ++i) c[i] = 0.0f;
            for (int kc = 0; kc < 2; ++kc) {
                const _Float16* krow = Kbase +
                    (size_t)(kb * 64 + j * 16 + ln) * D_H + kc * 32 + hl * 16;
                v16h b = *(const v16h*)krow;
                c = wmma_f16(aQ[kc], b, c);
            }
            sc[j] = c;
        }

        // scale + mask (mask shape N,1,1,S)
        float madd[4];
        for (int j = 0; j < 4; ++j)
            madd[j] = (mrow[kb * 64 + j * 16 + ln] == 0) ? -1.0e20f : 0.0f;
        for (int j = 0; j < 4; ++j)
            for (int rr = 0; rr < 8; ++rr)
                sc[j][rr] = sc[j][rr] * scale + madd[j];

        // ---- online softmax ---------------------------------------------
        float corr[8];
        for (int rr = 0; rr < 8; ++rr) {
            float v = fmaxf(fmaxf(sc[0][rr], sc[1][rr]), fmaxf(sc[2][rr], sc[3][rr]));
            v = fmaxf(v, __shfl_xor(v, 1));
            v = fmaxf(v, __shfl_xor(v, 2));
            v = fmaxf(v, __shfl_xor(v, 4));
            v = fmaxf(v, __shfl_xor(v, 8));
            const float mnew = fmaxf(m_r[rr], v);
            corr[rr] = __expf(m_r[rr] - mnew);
            m_r[rr] = mnew;
        }

        float rs[8];
        for (int rr = 0; rr < 8; ++rr) rs[rr] = 0.0f;
        for (int j = 0; j < 4; ++j) {
            for (int rr = 0; rr < 8; ++rr) {
                const float p = __expf(sc[j][rr] - m_r[rr]);
                rs[rr] += p;
                Pw[(rr + 8 * hl) * 64 + j * 16 + ln] = (_Float16)p;
            }
        }
        for (int rr = 0; rr < 8; ++rr) {
            float sum = rs[rr];
            sum += __shfl_xor(sum, 1);
            sum += __shfl_xor(sum, 2);
            sum += __shfl_xor(sum, 4);
            sum += __shfl_xor(sum, 8);
            l_r[rr] = l_r[rr] * corr[rr] + sum;
            for (int j = 0; j < 4; ++j) accO[j][rr] *= corr[rr];
        }

        // ---- O += P @ V (A from LDS b128, B = rows of transposed V) ------
        v16h aP[2];
        for (int lc = 0; lc < 2; ++lc) {
            const _Float16* prow = Pw + ln * 64 + lc * 32 + hl * 16;
            v8h lo = *(const v8h*)prow;           // 16B aligned -> ds_load_b128
            v8h hi = *(const v8h*)(prow + 8);
            for (int i = 0; i < 8; ++i) { aP[lc][i] = lo[i]; aP[lc][i + 8] = hi[i]; }
        }
        for (int j = 0; j < 4; ++j) {
            v8f c = accO[j];
            for (int lc = 0; lc < 2; ++lc) {
                const _Float16* vrow = VTbase +
                    (size_t)(j * 16 + ln) * S_LEN + kb * 64 + lc * 32 + hl * 16;
                v16h b = *(const v16h*)vrow;
                c = wmma_f16(aP[lc], b, c);
            }
            accO[j] = c;
        }
    }

    // epilogue: O / l -> stage in wave-private LDS -> coalesced b128 stores
    for (int j = 0; j < 4; ++j)
        for (int rr = 0; rr < 8; ++rr)
            Pw[(rr + 8 * hl) * 64 + j * 16 + ln] =
                (_Float16)(accO[j][rr] / l_r[rr]);

    for (int t = lane; t < 128; t += 32) {
        const int rowIdx = t >> 3;
        const int piece  = t & 7;
        const int q = qbase + rowIdx;
        *(v8h*)(attn_out + ((size_t)(n * S_LEN + q)) * E_DIM + h * D_H + piece * 8) =
            *(const v8h*)(Pw + rowIdx * 64 + piece * 8);
    }
}

// ---------------------------------------------------------------------------
// Stage 3: out = attn @ W_out^T + b_out  (M=8192, N=1024, K=1024), f32 out
// Epilogue staged through LDS for b128 f32 stores.
// ---------------------------------------------------------------------------
__global__ __launch_bounds__(256) void outproj_kernel(
        const _Float16* __restrict__ attn, const _Float16* __restrict__ Wout16,
        const float* __restrict__ bias, float* __restrict__ out) {
    __shared__ float fstage[8 * 16 * 64];   // 32 KB, one slab per wave
    const int wave = threadIdx.x >> 5;
    const int lane = threadIdx.x & 31;
    const int hl   = lane >> 4;
    const int ln   = lane & 15;
    const int rowBase = blockIdx.x * 128 + wave * 16;
    const int colBase = blockIdx.y * 64;
    float* fslab = fstage + wave * 16 * 64;

    const _Float16* arow = attn + (size_t)(rowBase + ln) * E_DIM;

    v8f acc[4];
    for (int j = 0; j < 4; ++j) for (int i = 0; i < 8; ++i) acc[j][i] = 0.0f;

    for (int kc = 0; kc < E_DIM / 32; ++kc) {
        v16h a;
        {
            const int base = kc * 32 + hl * 16;
            v8h lo = *(const v8h*)(arow + base);
            v8h hi = *(const v8h*)(arow + base + 8);
            for (int i = 0; i < 8; ++i) { a[i] = lo[i]; a[i + 8] = hi[i]; }
        }
        for (int j = 0; j < 4; ++j) {
            const _Float16* wp = Wout16 +
                (size_t)(colBase + j * 16 + ln) * E_DIM + kc * 32 + hl * 16;
            v16h b = *(const v16h*)wp;
            acc[j] = wmma_f16(a, b, acc[j]);
        }
    }

    for (int j = 0; j < 4; ++j) {
        const float bv = bias[colBase + j * 16 + ln];
        for (int rr = 0; rr < 8; ++rr)
            fslab[(rr + 8 * hl) * 64 + j * 16 + ln] = acc[j][rr] + bv;
    }

    // coalesced writeback: 16 rows x 256B f32, 16B chunks per lane
    for (int t = lane; t < 256; t += 32) {
        const int rowIdx = t >> 4;
        const int piece  = t & 15;
        *(v4f*)(out + (size_t)(rowBase + rowIdx) * E_DIM + colBase + piece * 4) =
            *(const v4f*)(fslab + rowIdx * 64 + piece * 4);
    }
}

// ---------------------------------------------------------------------------
// Host launcher
// ---------------------------------------------------------------------------
extern "C" void kernel_launch(void* const* d_in, const int* in_sizes, int n_in,
                              void* d_out, int out_size, void* d_ws, size_t ws_size,
                              hipStream_t stream) {
    (void)in_sizes; (void)n_in; (void)out_size; (void)ws_size;
    const float* values  = (const float*)d_in[0];
    const float* keys    = (const float*)d_in[1];
    const float* queries = (const float*)d_in[2];
    const int*   mask    = (const int*)d_in[3];
    const float* Wv      = (const float*)d_in[4];
    const float* Wk      = (const float*)d_in[5];
    const float* Wq      = (const float*)d_in[6];
    const float* Wout    = (const float*)d_in[7];
    const float* bout    = (const float*)d_in[8];
    float* out = (float*)d_out;

    // workspace carve-up (all f16 buffers), 256B aligned
    char* ws = (char*)d_ws;
    size_t off = 0;
    auto carve = [&](size_t bytes) -> void* {
        void* p = ws + off;
        off = (off + bytes + 255) & ~(size_t)255;
        return p;
    };
    const size_t projElems = (size_t)N_B * H_N * S_LEN * D_H;   // 8.39M
    _Float16* qp     = (_Float16*)carve(projElems * 2);
    _Float16* kp     = (_Float16*)carve(projElems * 2);
    _Float16* vpT    = (_Float16*)carve(projElems * 2);
    _Float16* attn   = (_Float16*)carve((size_t)N_B * S_LEN * E_DIM * 2);
    _Float16* Wq16   = (_Float16*)carve(D_H * D_H * 2);
    _Float16* Wk16   = (_Float16*)carve(D_H * D_H * 2);
    _Float16* Wv16   = (_Float16*)carve(D_H * D_H * 2);
    _Float16* Wout16 = (_Float16*)carve((size_t)E_DIM * E_DIM * 2);

    // Stage 0: weight conversion
    cvt_f32_f16_kernel<<<(D_H * D_H + 255) / 256, 256, 0, stream>>>(Wq, Wq16, D_H * D_H);
    cvt_f32_f16_kernel<<<(D_H * D_H + 255) / 256, 256, 0, stream>>>(Wk, Wk16, D_H * D_H);
    cvt_f32_f16_kernel<<<(D_H * D_H + 255) / 256, 256, 0, stream>>>(Wv, Wv16, D_H * D_H);
    cvt_f32_f16_kernel<<<(E_DIM * E_DIM + 255) / 256, 256, 0, stream>>>(Wout, Wout16, E_DIM * E_DIM);

    // Stage 1: per-head projections (131072 rows / 128 rows per WG)
    const int projBlocks = (N_B * S_LEN * H_N) / 128;
    proj_kernel<false><<<projBlocks, 256, 0, stream>>>(queries, Wq16, qp);
    proj_kernel<false><<<projBlocks, 256, 0, stream>>>(keys,    Wk16, kp);
    proj_kernel<true ><<<projBlocks, 256, 0, stream>>>(values,  Wv16, vpT);

    // Stage 2: fused flash attention: grid (qblocks=32, n*h=64), barrier-free
    attn_kernel<<<dim3(S_LEN / 64, N_B * H_N), 128, 0, stream>>>(qp, kp, vpT, mask, attn);

    // Stage 3: output projection + bias: grid (8192/128, 1024/64)
    outproj_kernel<<<dim3((N_B * S_LEN) / 128, E_DIM / 64), 256, 0, stream>>>(
        attn, Wout16, bout, out);
}
